// MutualCrossAttention_59631325938049
// MI455X (gfx1250) — matrix operand order repaired and checked by reading, not compiled, not run on hardware
//
#include <hip/hip_runtime.h>
#include <hip/hip_bf16.h>
#include <stdint.h>

// ---------------------------------------------------------------------------
// CDNA5 (gfx1250) WMMA types
// ---------------------------------------------------------------------------
typedef __attribute__((ext_vector_type(16))) __bf16 v16bf;
typedef __attribute__((ext_vector_type(8)))  float  v8f;

union BF16Frag { v16bf v; unsigned short u[16]; };

__device__ inline v8f zero8() {
    v8f z = {0.0f,0.0f,0.0f,0.0f,0.0f,0.0f,0.0f,0.0f};
    return z;
}

// float -> bf16 bits, round-to-nearest-even
__device__ inline unsigned short f2bf(float f) {
    union { float f; unsigned int u; } c; c.f = f;
    unsigned int u = c.u;
    return (unsigned short)((u + 0x7FFFu + ((u >> 16) & 1u)) >> 16);
}
__device__ inline unsigned short to_bf_bits(float f)          { return f2bf(f); }
__device__ inline unsigned short to_bf_bits(unsigned short u) { return u; }

// hardware-native reciprocal (v_rcp_f32): scores are rounded to bf16 right
// after the mask multiply, so ~1ulp rcp is lossless at the output precision.
__device__ inline float fast_rcp(float x) { return __builtin_amdgcn_rcpf(x); }

// Problem constants
#define TT   64      // time steps (== N == second T of output)
#define BB   8
#define NN   64
#define DD   384
#define HH   8
#define dd_  48
#define MROWS 32768          // T*B*N == T*B*T rows for every GEMM
#define INNER 196608         // B*N*D elements of LIF state

#define INV_SQRT_D 0.05103103630798287853f   // 384^-0.5

// ---------------------------------------------------------------------------
// Generic GEMM:  C[M][384] = A[M][384] * W[384][384]^T   (bf16 WMMA, f32 acc)
// Block = 128 threads (4 waves), 64x64 output tile, K-loop of 32.
// ---------------------------------------------------------------------------
template <typename TA>
__global__ __launch_bounds__(128)
void gemm_nt_kernel(const TA* __restrict__ A, const float* __restrict__ W,
                    float* __restrict__ C) {
    constexpr int K = DD;
    __shared__ unsigned short As[64][40];   // +8 pad vs 32 to spread banks
    __shared__ unsigned short Ws[64][40];   // stored as [n][k]

    const int tid  = threadIdx.x;
    const int lane = tid & 31;
    const int wave = tid >> 5;              // 0..3 -> 16-row strip
    const int lg   = lane >> 4;             // lane group (0/1)
    const int rowBase = blockIdx.x * 64;
    const int colBase = blockIdx.y * 64;

    v8f acc[4];
    #pragma unroll
    for (int t = 0; t < 4; ++t) acc[t] = zero8();

    for (int kt = 0; kt < K / 32; ++kt) {
        const int kBase = kt * 32;
        #pragma unroll
        for (int i = 0; i < 16; ++i) {
            int idx = tid + 128 * i;        // 0..2047 over 64x32 tile
            int r = idx >> 5, c = idx & 31;
            As[r][c] = to_bf_bits(A[(size_t)(rowBase + r) * K + kBase + c]);
            Ws[r][c] = f2bf(W[(size_t)(colBase + r) * K + kBase + c]);
        }
        __syncthreads();

        // A fragment: 16x32 strip for this wave (k = e + 16*lg, same order as B)
        BF16Frag af;
        {
            int r = wave * 16 + (lane & 15);
            #pragma unroll
            for (int e = 0; e < 16; ++e) {
                int k = e + 16 * lg;
                af.u[e] = As[r][k];
            }
        }
        #pragma unroll
        for (int nt = 0; nt < 4; ++nt) {
            // B fragment: column n per lane, K = e + 16*lg; B[k][n] = W[n][k]
            BF16Frag bfr;
            int n = nt * 16 + (lane & 15);
            int kOff = 16 * lg;
            #pragma unroll
            for (int e = 0; e < 16; ++e) bfr.u[e] = Ws[n][kOff + e];
            acc[nt] = __builtin_amdgcn_wmma_f32_16x16x32_bf16(
                false, af.v, false, bfr.v, (short)0, acc[nt], false, false);
        }
        __syncthreads();
    }

    // C/D layout: element v -> row = v + 8*lg (+strip), col = lane&15
    const int r0 = rowBase + wave * 16 + 8 * lg;
    const int cc = lane & 15;
    #pragma unroll
    for (int nt = 0; nt < 4; ++nt) {
        int col = colBase + nt * 16 + cc;
        #pragma unroll
        for (int v = 0; v < 8; ++v)
            C[(size_t)(r0 + v) * DD + col] = acc[nt][v];
    }
}

// ---------------------------------------------------------------------------
// BatchNorm statistics: column sums / sumsq over 32768 rows (coalesced)
// ---------------------------------------------------------------------------
__global__ void zero_stats(float* d_sum, float* d_sq) {
    d_sum[threadIdx.x] = 0.0f;
    d_sq[threadIdx.x]  = 0.0f;
}

__global__ __launch_bounds__(384)
void bn_stats_partial(const float* __restrict__ Y, float* d_sum, float* d_sq,
                      int rowsPerBlock) {
    const int c  = threadIdx.x;                 // 384 channels
    const int r0 = blockIdx.x * rowsPerBlock;
    float s = 0.0f, q = 0.0f;
    for (int r = r0; r < r0 + rowsPerBlock; ++r) {
        float v = Y[(size_t)r * DD + c];
        s += v; q += v * v;
    }
    atomicAdd(&d_sum[c], s);
    atomicAdd(&d_sq[c],  q);
}

__global__ void bn_finalize(const float* d_sum, const float* d_sq,
                            const float* __restrict__ gamma,
                            const float* __restrict__ beta,
                            float* d_scale, float* d_shift, float invM) {
    const int c = threadIdx.x;
    float mean = d_sum[c] * invM;
    float var  = d_sq[c] * invM - mean * mean;   // biased var (matches jnp.var)
    float sc   = gamma[c] * rsqrtf(var + 1e-5f);
    d_scale[c] = sc;
    d_shift[c] = beta[c] - mean * sc;
}

// ---------------------------------------------------------------------------
// Fused (optional BN-apply) + multi-step LIF over leading axis of 64.
// h = (v + y)/2 ; spike = (h>=1) ; v = (1-spike)*h   (tau=2, hard reset)
// ---------------------------------------------------------------------------
template <bool USE_BN, bool OUT_BF16>
__global__ __launch_bounds__(256)
void bn_lif_kernel(const float* __restrict__ X,
                   const float* __restrict__ scale,
                   const float* __restrict__ shift,
                   void* __restrict__ out) {
    const int idx = blockIdx.x * blockDim.x + threadIdx.x;
    if (idx >= INNER) return;
    float sc = 1.0f, sh = 0.0f;
    if constexpr (USE_BN) {
        int c = idx % DD;
        sc = scale[c];
        sh = shift[c];
    }
    float v = 0.0f;
    for (int t = 0; t < TT; ++t) {
        float y = X[(size_t)t * INNER + idx] * sc + sh;
        float h = 0.5f * (v + y);
        bool  s = (h >= 1.0f);
        if constexpr (OUT_BF16)
            ((unsigned short*)out)[(size_t)t * INNER + idx] = s ? 0x3F80u : 0u;
        else
            ((float*)out)[(size_t)t * INNER + idx] = s ? 1.0f : 0.0f;
        v = s ? 0.0f : h;
    }
}

// ---------------------------------------------------------------------------
// Attention per (x,b,h):  S = Q*K^T (64x64, Kdim 48 zero-padded to 64),
// S *= proximity mask, Z = S*V (64x48, Kdim 64), Z *= D^-0.5.
// Writes z in permuted (i,b,x,h,d) layout directly.
// ---------------------------------------------------------------------------
__global__ __launch_bounds__(128)
void attn_kernel(const unsigned short* __restrict__ qs,
                 const unsigned short* __restrict__ ks,
                 const unsigned short* __restrict__ vs,
                 float* __restrict__ zperm) {
    constexpr int LDW = 72;                 // 64 cols + pad
    __shared__ unsigned short Qs[64 * LDW];
    __shared__ unsigned short Ks[64 * LDW];
    __shared__ unsigned short Vs[64 * LDW];
    __shared__ unsigned short Ss[64 * LDW];

    const int x = blockIdx.x, b = blockIdx.y, h = blockIdx.z;
    const int tid = threadIdx.x, lane = tid & 31, wave = tid >> 5;
    const int lg = lane >> 4;
    const int colOff = h * dd_;

    // Stage Q/K/V tiles (bf16 spikes), zero-pad d: 48 -> 64
    #pragma unroll
    for (int i = 0; i < 32; ++i) {
        int idx = tid + 128 * i;            // 0..4095
        int r = idx >> 6, c = idx & 63;
        unsigned short qv = 0, kvv = 0, vv = 0;
        if (c < dd_) {
            size_t qrow = ((size_t)x * BB + b) * NN + r;    // qh[x,b,i=r,h,:]
            size_t krow = ((size_t)r * BB + b) * NN + x;    // kh/vh[j=r,b,x,h,:]
            qv  = qs[qrow * DD + colOff + c];
            kvv = ks[krow * DD + colOff + c];
            vv  = vs[krow * DD + colOff + c];
        }
        Qs[r * LDW + c] = qv;
        Ks[r * LDW + c] = kvv;
        Vs[r * LDW + c] = vv;
    }
    __syncthreads();

    // ---- S = Q * K^T ----
    v8f sacc[4];
    #pragma unroll
    for (int t = 0; t < 4; ++t) sacc[t] = zero8();
    const int rA = wave * 16 + (lane & 15);
    for (int kt = 0; kt < 2; ++kt) {
        BF16Frag af;
        #pragma unroll
        for (int e = 0; e < 16; ++e) {
            int k = kt * 32 + e + 16 * lg;
            af.u[e] = Qs[rA * LDW + k];
        }
        #pragma unroll
        for (int nt = 0; nt < 4; ++nt) {
            BF16Frag bfr;
            int n = nt * 16 + (lane & 15);
            int kOff = kt * 32 + 16 * lg;
            #pragma unroll
            for (int e = 0; e < 16; ++e) bfr.u[e] = Ks[n * LDW + kOff + e]; // B[k][j]=K[j][k]
            sacc[nt] = __builtin_amdgcn_wmma_f32_16x16x32_bf16(
                false, af.v, false, bfr.v, (short)0, sacc[nt], false, false);
        }
    }
    // proximity mask (v_rcp_f32, not IEEE div) on f32 fragments -> bf16 LDS
    #pragma unroll
    for (int nt = 0; nt < 4; ++nt) {
        #pragma unroll
        for (int v = 0; v < 8; ++v) {
            int i = wave * 16 + 8 * lg + v;
            int j = nt * 16 + (lane & 15);
            float m = fast_rcp(1.0f + fabsf((float)(i - j)));
            Ss[i * LDW + j] = f2bf(sacc[nt][v] * m);
        }
    }
    __syncthreads();

    // ---- Z = S * V ----
    v8f zacc[3];
    #pragma unroll
    for (int t = 0; t < 3; ++t) zacc[t] = zero8();
    for (int kt = 0; kt < 2; ++kt) {
        BF16Frag af;
        #pragma unroll
        for (int e = 0; e < 16; ++e) {
            int k = kt * 32 + e + 16 * lg;
            af.u[e] = Ss[rA * LDW + k];
        }
        #pragma unroll
        for (int nt = 0; nt < 3; ++nt) {
            BF16Frag bfr;
            int n = nt * 16 + (lane & 15);
            int kOff = kt * 32 + 16 * lg;
            #pragma unroll
            for (int e = 0; e < 16; ++e) bfr.u[e] = Vs[(kOff + e) * LDW + n]; // B[j][d]=V[j][d]
            zacc[nt] = __builtin_amdgcn_wmma_f32_16x16x32_bf16(
                false, af.v, false, bfr.v, (short)0, zacc[nt], false, false);
        }
    }
    #pragma unroll
    for (int nt = 0; nt < 3; ++nt) {
        #pragma unroll
        for (int v = 0; v < 8; ++v) {
            int i  = wave * 16 + 8 * lg + v;
            int dc = nt * 16 + (lane & 15);
            size_t o = (((size_t)i * BB + b) * TT + x) * DD + colOff + dc;
            zperm[o] = zacc[nt][v] * INV_SQRT_D;
        }
    }
}

// ---------------------------------------------------------------------------
// Host-side orchestration
// ---------------------------------------------------------------------------
extern "C" void kernel_launch(void* const* d_in, const int* in_sizes, int n_in,
                              void* d_out, int out_size, void* d_ws, size_t ws_size,
                              hipStream_t stream) {
    (void)in_sizes; (void)n_in; (void)out_size; (void)ws_size;
    const float* q  = (const float*)d_in[0];
    const float* kv = (const float*)d_in[1];
    const float* Wq = (const float*)d_in[2];
    const float* gq = (const float*)d_in[3];
    const float* bq = (const float*)d_in[4];
    const float* Wk = (const float*)d_in[5];
    const float* gk = (const float*)d_in[6];
    const float* bk = (const float*)d_in[7];
    const float* Wv = (const float*)d_in[8];
    const float* gv = (const float*)d_in[9];
    const float* bv = (const float*)d_in[10];
    const float* Wp = (const float*)d_in[11];
    const float* gp = (const float*)d_in[12];
    const float* bp = (const float*)d_in[13];

    // workspace layout (bytes)
    char* ws = (char*)d_ws;
    const size_t SZ_F32 = (size_t)MROWS * DD * sizeof(float);          // 50,331,648
    const size_t SZ_BF  = (size_t)MROWS * DD * sizeof(unsigned short); // 25,165,824
    float*          bufA  = (float*)(ws);                       // GEMM output
    float*          bufB  = (float*)(ws + SZ_F32);              // permuted z
    unsigned short* qs    = (unsigned short*)(ws + 2 * SZ_F32);
    unsigned short* ksb   = (unsigned short*)(ws + 2 * SZ_F32 + SZ_BF);
    unsigned short* vsb   = (unsigned short*)(ws + 2 * SZ_F32 + 2 * SZ_BF);
    float*          d_sum   = (float*)(ws + 2 * SZ_F32 + 3 * SZ_BF);
    float*          d_sq    = d_sum + 384;
    float*          d_scale = d_sum + 768;
    float*          d_shift = d_sum + 1152;

    const dim3 gGemm(MROWS / 64, DD / 64);   // 512 x 6
    const float invM = 1.0f / (float)MROWS;

    // --- three spiking branches: GEMM -> BN stats -> BN+LIF -> bf16 spikes
    auto branch = [&](const float* X, const float* W, const float* g,
                      const float* be, unsigned short* spk) {
        gemm_nt_kernel<float><<<gGemm, 128, 0, stream>>>(X, W, bufA);
        zero_stats<<<1, 384, 0, stream>>>(d_sum, d_sq);
        bn_stats_partial<<<64, 384, 0, stream>>>(bufA, d_sum, d_sq, MROWS / 64);
        bn_finalize<<<1, 384, 0, stream>>>(d_sum, d_sq, g, be, d_scale, d_shift, invM);
        bn_lif_kernel<true, true><<<INNER / 256, 256, 0, stream>>>(
            bufA, d_scale, d_shift, spk);
    };
    branch(q,  Wq, gq, bq, qs);
    branch(kv, Wk, gk, bk, ksb);
    branch(kv, Wv, gv, bv, vsb);

    // --- attention over 4096 (x,b,h) tiles, output already permuted
    attn_kernel<<<dim3(TT, BB, HH), 128, 0, stream>>>(qs, ksb, vsb, bufB);

    // --- attn LIF (no BN), spikes to bf16 (reuse qs buffer)
    bn_lif_kernel<false, true><<<INNER / 256, 256, 0, stream>>>(
        bufB, nullptr, nullptr, qs);

    // --- projection GEMM (bf16 spike A) -> BN -> LIF -> f32 output
    gemm_nt_kernel<unsigned short><<<gGemm, 128, 0, stream>>>(qs, Wp, bufA);
    zero_stats<<<1, 384, 0, stream>>>(d_sum, d_sq);
    bn_stats_partial<<<64, 384, 0, stream>>>(bufA, d_sum, d_sq, MROWS / 64);
    bn_finalize<<<1, 384, 0, stream>>>(d_sum, d_sq, gp, bp, d_scale, d_shift, invM);
    bn_lif_kernel<true, false><<<INNER / 256, 256, 0, stream>>>(
        bufA, d_scale, d_shift, d_out);
}
